// VectorQuantizerLayer_35390530520001
// MI455X (gfx1250) — compile-verified, matrix operand order checked by reading, not run on hardware
//
#include <hip/hip_runtime.h>

#define DIM 4096   // in/out features (reduction dim of the GEMM)
#define KQ  128    // codebook size (N dim of the GEMM)

typedef __attribute__((ext_vector_type(16))) __bf16 v16bf;
typedef __attribute__((ext_vector_type(8)))  __bf16 v8bf;
typedef __attribute__((ext_vector_type(8)))  float  v8f;

// ---------------------------------------------------------------------------
// Kernel 1: gathered^T = QV[assign[d], n] -> gT[n][d], split into bf16 hi/lo.
// Transposed layout makes the WMMA B-matrix per-lane loads contiguous in the
// reduction dimension. Output: 2 x 1 MB in d_ws.
// ---------------------------------------------------------------------------
__global__ __launch_bounds__(256) void vq_gather_split_kernel(
    const float* __restrict__ qv, const int* __restrict__ assign,
    unsigned short* __restrict__ gT_hi_u, unsigned short* __restrict__ gT_lo_u)
{
    const int d = blockIdx.x * blockDim.x + threadIdx.x;
    if (d >= DIM) return;
    __bf16* gT_hi = (__bf16*)gT_hi_u;
    __bf16* gT_lo = (__bf16*)gT_lo_u;
    const int a = assign[d];
    const float* row = qv + (size_t)a * KQ;
#pragma unroll 4
    for (int n = 0; n < KQ; ++n) {
        float  f = row[n];
        __bf16 h = (__bf16)f;                  // RNE truncation to bf16
        float  r = f - (float)h;               // residual
        gT_hi[(size_t)n * DIM + d] = h;
        gT_lo[(size_t)n * DIM + d] = (__bf16)r;
    }
}

__device__ __forceinline__ v8f wmma_bf16(v16bf a, v16bf b, v8f c) {
    return __builtin_amdgcn_wmma_f32_16x16x32_bf16(
        /*neg_a=*/false, a, /*neg_b=*/false, b,
        /*c_mod=*/(short)0, c, /*reuse_a=*/false, /*reuse_b=*/false);
}

// ---------------------------------------------------------------------------
// Kernel 2: fused GEMM (split-precision bf16 WMMA, fp32 accumulate) + column
// gather expansion. One block = 128 rows of x; 4 waves, each owning 32 rows
// x 128 cols (16 v8f accumulators). qm tile lives in 64 KB LDS, never in HBM.
// ---------------------------------------------------------------------------
__global__ __launch_bounds__(128) void vq_gemm_expand_kernel(
    const float* __restrict__ x, const int* __restrict__ assign,
    const float* __restrict__ bias,
    const unsigned short* __restrict__ gT_hi_u,
    const unsigned short* __restrict__ gT_lo_u,
    float* __restrict__ out)
{
    __shared__ float qm[128 * 128];            // 64 KB: qm tile [128 rows][128 cols]

    const __bf16* gT_hi = (const __bf16*)gT_hi_u;
    const __bf16* gT_lo = (const __bf16*)gT_lo_u;

    const int tid  = threadIdx.x;
    const int wave = tid >> 5;
    const int lane = tid & 31;
    const int l16  = lane & 15;
    const int half = lane >> 4;                // ISA 16-bit layout: lane<16 -> K 0-7 & 16-23
    const int c0   = half * 8;                 // lane>=16 -> K 8-15 & 24-31

    const size_t rowBase = (size_t)blockIdx.x * 128;

    // A-matrix: lane l16 holds row (base + l16) of each 16-row subtile.
    const float* xr0 = x + (rowBase + (size_t)wave * 32 +  0 + l16) * DIM;
    const float* xr1 = x + (rowBase + (size_t)wave * 32 + 16 + l16) * DIM;

    v8f acc[2][8];
#pragma unroll
    for (int s = 0; s < 2; ++s)
#pragma unroll
        for (int nt = 0; nt < 8; ++nt)
            acc[s][nt] = (v8f){0.f, 0.f, 0.f, 0.f, 0.f, 0.f, 0.f, 0.f};

    for (int d0 = 0; d0 < DIM; d0 += 32) {
        __builtin_prefetch(xr0 + d0 + 32, 0, 1);   // global_prefetch next A slice
        __builtin_prefetch(xr1 + d0 + 32, 0, 1);

        // ---- load A slice (fp32) and split to bf16 hi/lo in WMMA layout ----
        v16bf a_hi[2], a_lo[2];
#pragma unroll
        for (int s = 0; s < 2; ++s) {
            const float* xr = s ? xr1 : xr0;
            const float4* p0 = (const float4*)(xr + d0 + c0);        // K c0..c0+7
            const float4* p1 = (const float4*)(xr + d0 + c0 + 16);   // K c0+16..c0+23
            float4 f0 = p0[0], f1 = p0[1], f2 = p1[0], f3 = p1[1];
            float v[16] = { f0.x, f0.y, f0.z, f0.w,  f1.x, f1.y, f1.z, f1.w,
                            f2.x, f2.y, f2.z, f2.w,  f3.x, f3.y, f3.z, f3.w };
#pragma unroll
            for (int e = 0; e < 16; ++e) {
                __bf16 h = (__bf16)v[e];
                a_hi[s][e] = h;
                a_lo[s][e] = (__bf16)(v[e] - (float)h);
            }
        }

        // ---- 8 column tiles: B loads + 6 WMMAs each (2 subtiles x 3 terms) --
#pragma unroll
        for (int nt = 0; nt < 8; ++nt) {
            const __bf16* bh = gT_hi + (size_t)(nt * 16 + l16) * DIM + d0 + c0;
            const __bf16* bl = gT_lo + (size_t)(nt * 16 + l16) * DIM + d0 + c0;
            v8bf h0 = *(const v8bf*)bh;
            v8bf h1 = *(const v8bf*)(bh + 16);
            v8bf l0 = *(const v8bf*)bl;
            v8bf l1 = *(const v8bf*)(bl + 16);
            v16bf b_hi = __builtin_shufflevector(h0, h1, 0,1,2,3,4,5,6,7,8,9,10,11,12,13,14,15);
            v16bf b_lo = __builtin_shufflevector(l0, l1, 0,1,2,3,4,5,6,7,8,9,10,11,12,13,14,15);

            acc[0][nt] = wmma_bf16(a_hi[0], b_hi, acc[0][nt]);
            acc[0][nt] = wmma_bf16(a_hi[0], b_lo, acc[0][nt]);
            acc[0][nt] = wmma_bf16(a_lo[0], b_hi, acc[0][nt]);
            acc[1][nt] = wmma_bf16(a_hi[1], b_hi, acc[1][nt]);
            acc[1][nt] = wmma_bf16(a_hi[1], b_lo, acc[1][nt]);
            acc[1][nt] = wmma_bf16(a_lo[1], b_hi, acc[1][nt]);
        }
    }

    // ---- spill accumulators to LDS per documented C/D layout ----
    // (VGPR j, lane l) holds row j + 8*(l>=16), col l&15 of the 16x16 tile.
#pragma unroll
    for (int s = 0; s < 2; ++s)
#pragma unroll
        for (int nt = 0; nt < 8; ++nt)
#pragma unroll
            for (int j = 0; j < 8; ++j) {
                int r = wave * 32 + s * 16 + half * 8 + j;
                qm[r * 128 + nt * 16 + l16] = acc[s][nt][j];
            }

    __syncthreads();

    // ---- fused expansion: out[t][d] = qm[t][assign[d]] + bias[d] ----
    // Threads cover 512 consecutive cols per step -> fully coalesced b128 stores.
    const int dl = tid * 4;
#pragma unroll 1
    for (int g = 0; g < 8; ++g) {
        const int d = g * 512 + dl;
        const int4   a4 = *(const int4*)(assign + d);
        const float4 b4 = *(const float4*)(bias + d);
        float* o = out + rowBase * DIM + d;
#pragma unroll 4
        for (int t = 0; t < 128; ++t) {
            const float* q = qm + t * 128;
            float4 v;
            v.x = q[a4.x] + b4.x;
            v.y = q[a4.y] + b4.y;
            v.z = q[a4.z] + b4.z;
            v.w = q[a4.w] + b4.w;
            *(float4*)(o + (size_t)t * DIM) = v;
        }
    }
}

// ---------------------------------------------------------------------------
extern "C" void kernel_launch(void* const* d_in, const int* in_sizes, int n_in,
                              void* d_out, int out_size, void* d_ws, size_t ws_size,
                              hipStream_t stream) {
    const float* x      = (const float*)d_in[0];   // [B,S,D] fp32
    const float* qv     = (const float*)d_in[1];   // [D,K]   fp32
    const int*   assign = (const int*)  d_in[2];   // [D]     int32
    const float* bias   = (const float*)d_in[3];   // [D]     fp32
    float*       out    = (float*)d_out;

    unsigned short* gT_hi = (unsigned short*)d_ws;                     // 1 MB
    unsigned short* gT_lo = gT_hi + (size_t)KQ * DIM;                  // 1 MB

    const int T = in_sizes[0] / DIM;               // 16384 rows

    vq_gather_split_kernel<<<(DIM + 255) / 256, 256, 0, stream>>>(
        qv, assign, gT_hi, gT_lo);
    vq_gemm_expand_kernel<<<T / 128, 128, 0, stream>>>(
        x, assign, bias, gT_hi, gT_lo, out);
}
